// RATEncoderLayer_22170621182357
// MI455X (gfx1250) — matrix operand order repaired and checked by reading, not compiled, not run on hardware
//
#include <hip/hip_runtime.h>
#include <hip/hip_bf16.h>

// ---------------------------------------------------------------------------
// RAT encoder layer for gfx1250 (MI455X), wave32 + WMMA f32_16x16x32_f16,
// with CDNA5 async global->LDS tile DMA (GLOBAL_LOAD_ASYNC_TO_LDS_B128).
//
//   xn  = LN1(x)                         -> f16
//   q/k/v = xn @ Wq/Wk/Wv + b            (WMMA GEMM, f16 out)
//   scores[h] = q_h @ k_h^T              (WMMA GEMM, BT mode, f32 out)
//   qrel[h,q,r] = q_h[q,:] . relk[r,:]   (small kernel)
//   softmax fused with +qrel[rel[q,k]] gather, *1/8, mask; also buckets
//       attn_rel[h,q,r] = sum_{rel==r} attn  (LDS atomics)
//   ctx[h] = attn_h @ v_h                (WMMA GEMM) ; ctx += attn_rel @ relv
//   x1  = x + ctx @ Wo + bo              (WMMA GEMM, residual epilogue)
//   out = x1 + relu(LN2(x1) @ W1 + b1) @ W2 + b2   (two WMMA GEMMs)
//
// GEMM assumptions (hold for every call site in this layer):
//   M % 128 == 0, K % 32 == 0, N % 16 == 0; all pointers/strides 16B-aligned.
// ---------------------------------------------------------------------------

typedef __attribute__((ext_vector_type(16))) _Float16 v16h;
typedef __attribute__((ext_vector_type(4)))  _Float16 v4h;
typedef __attribute__((ext_vector_type(8)))  float    v8f;

#define TS  1024   // S
#define TD  512    // D
#define TH  8      // H
#define TDK 64     // DK
#define TDFF 2048  // DFF
#define TNREL 51

#define BM 128
#define BN 128
#define BK 32
#define BKP 40     // padded LDS K-stride (halves); 40*2=80B, 16B-aligned rows

// Async DMA of 16 bytes per lane: global -> LDS, tracked by ASYNCcnt.
// LDS operand: flat-address low 32 bits == LDS byte offset (ISA 10.2 aperture).
__device__ __forceinline__ void async_b128(const void* gsrc, void* ldst) {
  unsigned loff = (unsigned)(unsigned long long)ldst;
  unsigned long long ga = (unsigned long long)gsrc;
  asm volatile("global_load_async_to_lds_b128 %0, %1, off"
               :: "v"(loff), "v"(ga) : "memory");
}
__device__ __forceinline__ void wait_async0() {
  asm volatile("s_wait_asynccnt 0x0" ::: "memory");
}

// ---------------------------------------------------------------------------
// Generic tiled WMMA GEMM: C = op(A[MxK] * B) with fused epilogue.
//   bTrans==0: B is [K x N] row-major (ldb)   -> C = A*B
//   bTrans==1: B is [N x K] row-major (ldb)   -> C = A*B^T  (requires N%128==0)
// Per-head batching via blockIdx.z and element strides aHS/bHS/cHS.
// ---------------------------------------------------------------------------
__global__ __launch_bounds__(256) void rat_wmma_gemm(
    const _Float16* __restrict__ A, int lda, long aHS,
    const _Float16* __restrict__ B, int ldb, long bHS, int bTrans,
    float* __restrict__ Cf32, _Float16* __restrict__ Cf16, int ldc, long cHS,
    const float* __restrict__ bias, const float* __restrict__ resid, int doRelu,
    int M, int N, int K)
{
  __shared__ _Float16 As[BM][BKP];
  __shared__ _Float16 Bs[BN][BKP];  // always stored [n][k]

  const int t    = threadIdx.x;
  const int lane = t & 31;
  const int wave = t >> 5;
  const int wm   = wave & 3;   // 4 waves along M, 32 rows each
  const int wn   = wave >> 2;  // 2 waves along N, 64 cols each
  const int m0   = blockIdx.y * BM;
  const int n0   = blockIdx.x * BN;
  const int head = blockIdx.z;

  A += (long)head * aHS;
  B += (long)head * bHS;
  if (Cf32) Cf32 += (long)head * cHS;
  if (Cf16) Cf16 += (long)head * cHS;

  const int lrow = lane & 15;
  const int lhi  = lane >> 4;

  v8f acc[2][4] = {};

  // per-thread tile-load coordinates (fixed across k-steps)
  const int arow = t >> 1;            // A/BT: row within tile
  const int aseg = (t & 1) << 4;      // A/BT: 16-half segment
  const int krow = t >> 3;            // BN: k-row within tile
  const int segn = (t & 7) << 4;      // BN: 16-col segment
  const bool bnIn = (n0 + segn + 16) <= N;  // uniform full/empty segment

  for (int k0 = 0; k0 < K; k0 += BK) {
    // ---- A tile: [BM][BK], 32B per thread via async LDS DMA
    {
      const _Float16* src = A + (long)(m0 + arow) * lda + k0 + aseg;
      async_b128(src,     &As[arow][aseg]);
      async_b128(src + 8, &As[arow][aseg + 8]);
    }
    // ---- B tile -> Bs[n][k]
    if (bTrans) {
      const _Float16* src = B + (long)(n0 + arow) * ldb + k0 + aseg;
      async_b128(src,     &Bs[arow][aseg]);
      async_b128(src + 8, &Bs[arow][aseg + 8]);
    } else {
      const _Float16* src = B + (long)(k0 + krow) * ldb + n0 + segn;
      _Float16 tmp[16];
      if (bnIn) {
        if (k0 + BK < K) __builtin_prefetch(src + (long)BK * ldb, 0, 1);
        float4 b0 = ((const float4*)src)[0];
        float4 b1 = ((const float4*)src)[1];
        *(float4*)&tmp[0] = b0;
        *(float4*)&tmp[8] = b1;
      } else {
#pragma unroll
        for (int j = 0; j < 16; ++j) tmp[j] = (_Float16)0.f;
      }
#pragma unroll
      for (int j = 0; j < 16; ++j) Bs[segn + j][krow] = tmp[j];  // transpose
    }
    wait_async0();
    __syncthreads();

    // ---- fragments (CDNA5 16-bit A 16x32 layout; B mirrored over N)
    v16h af[2];
#pragma unroll
    for (int mi = 0; mi < 2; ++mi) {
      int row = wm * 32 + mi * 16 + lrow;
      int kb  = lhi * 8;
#pragma unroll
      for (int j = 0; j < 8; ++j) {
        af[mi][j]     = As[row][kb + j];
        af[mi][j + 8] = As[row][kb + 16 + j];
      }
    }
    v16h bf[4];
#pragma unroll
    for (int ni = 0; ni < 4; ++ni) {
      int col = wn * 64 + ni * 16 + lrow;
      int kb  = lhi * 8;
#pragma unroll
      for (int j = 0; j < 8; ++j) {
        bf[ni][j]     = Bs[col][kb + j];
        bf[ni][j + 8] = Bs[col][kb + 16 + j];
      }
    }
#pragma unroll
    for (int mi = 0; mi < 2; ++mi)
#pragma unroll
      for (int ni = 0; ni < 4; ++ni)
        acc[mi][ni] = __builtin_amdgcn_wmma_f32_16x16x32_f16(
            false, af[mi], false, bf[ni], (short)0, acc[mi][ni], false, false);
    __syncthreads();
  }

  // ---- epilogue: D layout: VGPR r -> M = r + 8*(lane>=16), N = lane&15
#pragma unroll
  for (int mi = 0; mi < 2; ++mi) {
#pragma unroll
    for (int ni = 0; ni < 4; ++ni) {
      int col = n0 + wn * 64 + ni * 16 + lrow;
      if (col >= N) continue;
      float bv = bias ? bias[col] : 0.f;
#pragma unroll
      for (int r = 0; r < 8; ++r) {
        int grow = m0 + wm * 32 + mi * 16 + r + lhi * 8;
        float v = acc[mi][ni][r] + bv;
        if (resid) v += resid[(long)grow * ldc + col];
        if (doRelu) v = v > 0.f ? v : 0.f;
        long off = (long)grow * ldc + col;
        if (Cf32) Cf32[off] = v;
        if (Cf16) Cf16[off] = (_Float16)v;
      }
    }
  }
}

// ---------------------------------------------------------------------------
// LayerNorm over last dim (D==512 fixed) -> f16 output
// ---------------------------------------------------------------------------
__global__ __launch_bounds__(256) void rat_ln_f16(
    const float* __restrict__ x, const float* __restrict__ g,
    const float* __restrict__ b, _Float16* __restrict__ out)
{
  int row = blockIdx.x;
  int t   = threadIdx.x;
  const float* xr = x + (long)row * TD;
  float e0 = xr[t], e1 = xr[t + 256];
  __shared__ float rs[256], rq[256];
  rs[t] = e0 + e1;
  rq[t] = e0 * e0 + e1 * e1;
  __syncthreads();
  for (int s = 128; s > 0; s >>= 1) {
    if (t < s) { rs[t] += rs[t + s]; rq[t] += rq[t + s]; }
    __syncthreads();
  }
  float mean = rs[0] * (1.f / TD);
  float var  = rq[0] * (1.f / TD) - mean * mean;
  float inv  = rsqrtf(var + 1e-5f);
  out[(long)row * TD + t]       = (_Float16)((e0 - mean) * inv * g[t] + b[t]);
  out[(long)row * TD + t + 256] = (_Float16)((e1 - mean) * inv * g[t + 256] + b[t + 256]);
}

// vectorized f32 -> f16 (n must be a multiple of 4; true for all call sites)
__global__ void rat_f32_to_f16(const float* __restrict__ in,
                               _Float16* __restrict__ out, long n4)
{
  long i = (long)blockIdx.x * blockDim.x + threadIdx.x;
  if (i < n4) {
    float4 v = ((const float4*)in)[i];
    v4h h;
    h[0] = (_Float16)v.x; h[1] = (_Float16)v.y;
    h[2] = (_Float16)v.z; h[3] = (_Float16)v.w;
    ((v4h*)out)[i] = h;
  }
}

// qrel[h,q,r] = q_h[q,:] . rel_k_emb[r,:]
__global__ __launch_bounds__(64) void rat_qrel(
    const _Float16* __restrict__ q16, const float* __restrict__ relk,
    float* __restrict__ qrel)
{
  int qi = blockIdx.x, h = blockIdx.y, r = threadIdx.x;
  if (r >= TNREL) return;
  const _Float16* qr = q16 + (long)qi * TD + h * TDK;
  const float*    rk = relk + (long)r * TDK;
  float s = 0.f;
#pragma unroll 8
  for (int j = 0; j < TDK; ++j) s += (float)qr[j] * rk[j];
  qrel[((long)h * TS + qi) * TNREL + r] = s;
}

// fused: bias gather + scale + mask + softmax + f16 store + relation buckets
__global__ __launch_bounds__(256) void rat_softmax(
    const float* __restrict__ scores, const int* __restrict__ relation,
    const int* __restrict__ mask, const float* __restrict__ qrel,
    _Float16* __restrict__ attn, float* __restrict__ attnrel)
{
  int qi = blockIdx.x, h = blockIdx.y, t = threadIdx.x;
  const float* srow = scores + ((long)h * TS + qi) * TS;
  const int*   rrow = relation + (long)qi * TS;
  const int*   mrow = mask + (long)qi * TS;
  const float* qrow = qrel + ((long)h * TS + qi) * TNREL;

  float vals[4]; int rels[4];
  float lmax = -3.0e38f;
#pragma unroll
  for (int i = 0; i < 4; ++i) {
    int k = t + i * 256;
    int r = rrow[k]; rels[i] = r;
    float sv = (srow[k] + qrow[r]) * 0.125f;   // 1/sqrt(64)
    if (mrow[k] == 0) sv = -1e9f;
    vals[i] = sv;
    lmax = fmaxf(lmax, sv);
  }
  __shared__ float red[256];
  __shared__ float buckets[64];
  red[t] = lmax;
  __syncthreads();
  for (int s = 128; s > 0; s >>= 1) {
    if (t < s) red[t] = fmaxf(red[t], red[t + s]);
    __syncthreads();
  }
  float m = red[0];
  __syncthreads();
  float lsum = 0.f;
#pragma unroll
  for (int i = 0; i < 4; ++i) { vals[i] = __expf(vals[i] - m); lsum += vals[i]; }
  red[t] = lsum;
  if (t < 64) buckets[t] = 0.f;
  __syncthreads();
  for (int s = 128; s > 0; s >>= 1) {
    if (t < s) red[t] += red[t + s];
    __syncthreads();
  }
  float inv = 1.f / red[0];
  _Float16* arow = attn + ((long)h * TS + qi) * TS;
#pragma unroll
  for (int i = 0; i < 4; ++i) {
    int k = t + i * 256;
    float p = vals[i] * inv;
    arow[k] = (_Float16)p;
    atomicAdd(&buckets[rels[i]], p);
  }
  __syncthreads();
  if (t < TNREL) attnrel[((long)h * TS + qi) * TNREL + t] = buckets[t];
}

// ctx[q, h*64+d] += attn_rel[h,q,:] @ rel_v_emb[:,d]
__global__ __launch_bounds__(64) void rat_relv(
    const float* __restrict__ attnrel, const float* __restrict__ relv,
    float* __restrict__ ctx)
{
  int qi = blockIdx.x, h = blockIdx.y, d = threadIdx.x;
  const float* ar = attnrel + ((long)h * TS + qi) * TNREL;
  float s = 0.f;
#pragma unroll
  for (int r = 0; r < TNREL; ++r) s += ar[r] * relv[(long)r * TDK + d];
  ctx[(long)qi * TD + h * TDK + d] += s;
}

// ---------------------------------------------------------------------------
extern "C" void kernel_launch(void* const* d_in, const int* in_sizes, int n_in,
                              void* d_out, int out_size, void* d_ws, size_t ws_size,
                              hipStream_t stream) {
  const float* x        = (const float*)d_in[0];
  const int*   relation = (const int*)d_in[1];
  const int*   mask     = (const int*)d_in[2];
  const float* wq = (const float*)d_in[3];
  const float* bq = (const float*)d_in[4];
  const float* wk = (const float*)d_in[5];
  const float* bk = (const float*)d_in[6];
  const float* wv = (const float*)d_in[7];
  const float* bv = (const float*)d_in[8];
  const float* wo = (const float*)d_in[9];
  const float* bo = (const float*)d_in[10];
  const float* rel_k_emb = (const float*)d_in[11];
  const float* rel_v_emb = (const float*)d_in[12];
  const float* ln1_g = (const float*)d_in[13];
  const float* ln1_b = (const float*)d_in[14];
  const float* ln2_g = (const float*)d_in[15];
  const float* ln2_b = (const float*)d_in[16];
  const float* w1 = (const float*)d_in[17];
  const float* b1 = (const float*)d_in[18];
  const float* w2 = (const float*)d_in[19];
  const float* b2 = (const float*)d_in[20];
  float* out = (float*)d_out;

  // ---- workspace carve-up
  char* w = (char*)d_ws;
  size_t off = 0;
  auto alloc = [&](size_t bytes) {
    void* p = w + off;
    off = (off + bytes + 255) & ~(size_t)255;
    return p;
  };
  _Float16* xn16  = (_Float16*)alloc((size_t)TS * TD * 2);
  _Float16* q16   = (_Float16*)alloc((size_t)TS * TD * 2);
  _Float16* k16   = (_Float16*)alloc((size_t)TS * TD * 2);
  _Float16* v16   = (_Float16*)alloc((size_t)TS * TD * 2);
  _Float16* wq16  = (_Float16*)alloc((size_t)TD * TD * 2);
  _Float16* wk16  = (_Float16*)alloc((size_t)TD * TD * 2);
  _Float16* wv16  = (_Float16*)alloc((size_t)TD * TD * 2);
  _Float16* wo16  = (_Float16*)alloc((size_t)TD * TD * 2);
  _Float16* w116  = (_Float16*)alloc((size_t)TD * TDFF * 2);
  _Float16* w216  = (_Float16*)alloc((size_t)TDFF * TD * 2);
  float*    scores  = (float*)alloc((size_t)TH * TS * TS * 4);
  _Float16* attn16  = (_Float16*)alloc((size_t)TH * TS * TS * 2);
  float*    qrel    = (float*)alloc((size_t)TH * TS * TNREL * 4);
  float*    attnrel = (float*)alloc((size_t)TH * TS * TNREL * 4);
  float*    ctx32   = (float*)alloc((size_t)TS * TD * 4);
  _Float16* ctx16   = (_Float16*)alloc((size_t)TS * TD * 2);
  float*    x1      = (float*)alloc((size_t)TS * TD * 4);
  _Float16* xn216   = (_Float16*)alloc((size_t)TS * TD * 2);
  _Float16* h16     = (_Float16*)alloc((size_t)TS * TDFF * 2);

  // ---- weight conversions f32 -> f16 (vectorized x4)
  auto conv = [&](const float* src, _Float16* dst, long n) {
    long n4 = n >> 2;
    rat_f32_to_f16<<<dim3((n4 + 255) / 256), dim3(256), 0, stream>>>(src, dst, n4);
  };
  conv(wq, wq16, (long)TD * TD);
  conv(wk, wk16, (long)TD * TD);
  conv(wv, wv16, (long)TD * TD);
  conv(wo, wo16, (long)TD * TD);
  conv(w1, w116, (long)TD * TDFF);
  conv(w2, w216, (long)TDFF * TD);

  auto gemm = [&](const _Float16* A, int lda, long aHS,
                  const _Float16* B, int ldb, long bHS, int bT,
                  float* Cf, _Float16* Ch, int ldc, long cHS,
                  const float* bias, const float* resid, int relu,
                  int M, int N, int K, int heads) {
    dim3 g((N + BN - 1) / BN, (M + BM - 1) / BM, heads);
    rat_wmma_gemm<<<g, dim3(256), 0, stream>>>(A, lda, aHS, B, ldb, bHS, bT,
                                               Cf, Ch, ldc, cHS, bias, resid,
                                               relu, M, N, K);
  };

  // ---- sublayer 0
  rat_ln_f16<<<dim3(TS), dim3(256), 0, stream>>>(x, ln1_g, ln1_b, xn16);

  gemm(xn16, TD, 0, wq16, TD, 0, 0, nullptr, q16, TD, 0, bq, nullptr, 0, TS, TD, TD, 1);
  gemm(xn16, TD, 0, wk16, TD, 0, 0, nullptr, k16, TD, 0, bk, nullptr, 0, TS, TD, TD, 1);
  gemm(xn16, TD, 0, wv16, TD, 0, 0, nullptr, v16, TD, 0, bv, nullptr, 0, TS, TD, TD, 1);

  // scores[h] = q_h @ k_h^T   (per-head batched via z)
  gemm(q16, TD, TDK, k16, TD, TDK, 1, scores, nullptr, TS, (long)TS * TS,
       nullptr, nullptr, 0, TS, TS, TDK, TH);

  rat_qrel<<<dim3(TS, TH), dim3(64), 0, stream>>>(q16, rel_k_emb, qrel);

  rat_softmax<<<dim3(TS, TH), dim3(256), 0, stream>>>(scores, relation, mask,
                                                      qrel, attn16, attnrel);

  // ctx[h] = attn_h @ v_h
  gemm(attn16, TS, (long)TS * TS, v16, TD, TDK, 0, ctx32, nullptr, TD, TDK,
       nullptr, nullptr, 0, TS, TDK, TS, TH);

  rat_relv<<<dim3(TS, TH), dim3(64), 0, stream>>>(attnrel, rel_v_emb, ctx32);

  conv(ctx32, ctx16, (long)TS * TD);

  // x1 = x + ctx @ Wo + bo
  gemm(ctx16, TD, 0, wo16, TD, 0, 0, x1, nullptr, TD, 0, bo, x, 0, TS, TD, TD, 1);

  // ---- sublayer 1
  rat_ln_f16<<<dim3(TS), dim3(256), 0, stream>>>(x1, ln2_g, ln2_b, xn216);

  gemm(xn216, TD, 0, w116, TDFF, 0, 0, nullptr, h16, TDFF, 0, b1, nullptr, 1,
       TS, TDFF, TD, 1);

  gemm(h16, TDFF, 0, w216, TD, 0, 0, out, nullptr, TD, 0, b2, x1, 0,
       TS, TD, TDFF, 1);
}
